// AFNO1D_46660524703767
// MI455X (gfx1250) — matrix operand order repaired and checked by reading, not compiled, not run on hardware
//
#include <hip/hip_runtime.h>
#include <hip/hip_bf16.h>

// ---------------------------------------------------------------------------
// AFNO1D collapsed to per-block GEMM chain for MI455X (gfx1250, wave32, WMMA)
//   xf   = x @ H1024                                   (bf16 WMMA, f32 accum)
//   o1   = relu( xf[b]@A1_blk + xf[flip(b)]@B1_blk + b1[0] )
//   z    = softshrink( o1[b]@A2_blk + o1[flip(b)]@B2_blk + b2[0] )
//   out  = z @ (H128 / 2^24) + x        (per 128-wide block)
// B-side matrices are pre-swizzled into a WMMA B-fragment lane layout with K = 16*(lane/16) + e
// (32B contiguous per lane per 32x16 tile) and staged once per workgroup into
// LDS; all 8 waves of a workgroup share the panel across 8 row-tiles each.
// All B fragments of a K-step are prefetched into registers before the WMMA
// burst so ds_load latency overlaps matrix issue instead of serializing.
// ---------------------------------------------------------------------------

typedef __attribute__((ext_vector_type(16))) __bf16 v16bf;
typedef __attribute__((ext_vector_type(8)))  __bf16 v8bf;
typedef __attribute__((ext_vector_type(8)))  float  v8f;

static constexpr int   Btot   = 4;
static constexpr int   Nseq   = 4096;
static constexpr int   HID    = 1024;
static constexpr int   NB     = 8;
static constexpr int   BS     = 128;
static constexpr int   ROWS   = Btot * Nseq;              // 16384
static constexpr float INV_NUMEL = 1.0f / 16777216.0f;    // 1 / (4*4096*8*128)
static constexpr float CHALF     = 0.5f / 16777216.0f;
static constexpr float LAMBDA    = 0.01f;

// ---------------------------- WMMA fragment helpers ------------------------
// A 16x32 bf16 layout: row M=lane&15, elements = 16 contiguous K:
// K in [16h, 16h+16), h = lane>>4 (matches packed B layout).
__device__ __forceinline__ v16bf load_a_frag(const __bf16* __restrict__ X, int ld,
                                             int row0, int k0, int lane) {
  const int half = (lane >> 4) & 1, r = lane & 15;
  const __bf16* base = X + (size_t)(row0 + r) * ld + k0 + 16 * half;
  const v8bf lo = *(const v8bf*)(base);
  const v8bf hi = *(const v8bf*)(base + 8);
  return __builtin_shufflevector(lo, hi, 0, 1, 2, 3, 4, 5, 6, 7,
                                 8, 9, 10, 11, 12, 13, 14, 15);
}

// Pre-swizzled B fragment from LDS: 16 contiguous bf16 per lane per tile.
__device__ __forceinline__ v16bf load_b_lds(const __bf16* Wp, int tileIdx, int lane) {
  return *(const v16bf*)(Wp + (((size_t)tileIdx * 32 + lane) << 4));
}

// Packed offset of element (k, n) of a K x N matrix, KT = K/32 tiles along K.
__device__ __forceinline__ int pack_off(int k, int n, int KT) {
  const int tk = k >> 5, tn = n >> 4;
  const int lane = (((k >> 4) & 1) << 4) | (n & 15);
  const int e = k & 15;
  return (((tn * KT + tk) << 9) | (lane << 4) | e);
}

__device__ __forceinline__ v8f wmma_bf16(v16bf a, v16bf b, v8f c) {
  // (neg_a, A, neg_b, B, c_mod, C, reuse_a, reuse_b)
  return __builtin_amdgcn_wmma_f32_16x16x32_bf16(false, a, false, b, (short)0, c,
                                                 false, false);
}

__device__ __forceinline__ float casf(int p, float scale) {  // cas(p*scale)
  float s, c;
  __sincosf((float)p * scale, &s, &c);
  return c + s;
}

// b128 memcpy global -> LDS (compiler: global_load_b128 + ds_store_b128)
__device__ __forceinline__ void stage_lds(void* dst, const void* src, int bytes,
                                          int tid, int nthreads) {
  uint4* d = (uint4*)dst;
  const uint4* s = (const uint4*)src;
  const int n = bytes >> 4;
  for (int i = tid; i < n; i += nthreads) d[i] = s[i];
}

// ------------------------------- setup kernels -----------------------------
// H1024 (packed bf16), H128 (row-major f32 for setup math), Hf = H128/2^24
// (packed bf16). Exact modular reduction: cas(2*pi*((k*n) mod L)/L).
__global__ void k_build_h(__bf16* __restrict__ H1024p, float* __restrict__ H128f,
                          __bf16* __restrict__ Hfp) {
  const size_t n = (size_t)1024 * 1024;
  for (size_t i = (size_t)blockIdx.x * blockDim.x + threadIdx.x; i < n;
       i += (size_t)gridDim.x * blockDim.x) {
    // invert packed layout for K=N=1024, KT=32
    const int e = (int)i & 15, lane = ((int)i >> 4) & 31, tile = (int)i >> 9;
    const int tk = tile & 31, tn = tile >> 5;
    const int k = tk * 32 + ((lane >> 4) << 4) + e;
    const int m = tn * 16 + (lane & 15);
    H1024p[i] = (__bf16)casf((k * m) & 1023, 6.135923151542565e-3f);  // 2pi/1024
    if (i < (size_t)BS * BS) {
      // row-major f32 H128
      const int k2 = (int)(i >> 7), m2 = (int)(i & 127);
      H128f[i] = casf((k2 * m2) & 127, 4.908738521234052e-2f);        // 2pi/128
      // packed Hf for K=N=128, KT=4
      const int tile2 = (int)i >> 9;
      const int tk2 = tile2 & 3, tn2 = tile2 >> 2;
      const int k3 = tk2 * 32 + ((lane >> 4) << 4) + e;
      const int m3 = tn2 * 16 + (lane & 15);
      Hfp[i] = (__bf16)(casf((k3 * m3) & 127, 4.908738521234052e-2f) * INV_NUMEL);
    }
  }
}

// S[blk] = w[0][blk] + w[1][blk]  (one layer per launch)
__global__ void k_sumw(const float* __restrict__ w, float* __restrict__ S) {
  const int i = blockIdx.x * blockDim.x + threadIdx.x;
  if (i < NB * BS * BS) S[i] = w[i] + w[NB * BS * BS + i];
}

// T[l,blk] = S[l,blk] @ H128
__global__ void k_matT(const float* __restrict__ S, const float* __restrict__ H,
                       float* __restrict__ T) {
  const int idx = blockIdx.x * blockDim.x + threadIdx.x;
  if (idx >= 2 * NB * BS * BS) return;
  const int o = idx & 127, m = (idx >> 7) & 127, mb = idx >> 14;
  const float* s = S + ((size_t)mb << 14) + (m << 7);
  float acc = 0.f;
  for (int i = 0; i < BS; ++i) acc += s[i] * H[i * BS + o];
  T[idx] = acc;
}

// B_blk = c*((T - S[fblk]) @ H)  (stored packed);  UH = (T + S[fblk]) @ H
__global__ void k_vh(const float* __restrict__ T, const float* __restrict__ S,
                     const float* __restrict__ H, float* __restrict__ UH,
                     __bf16* __restrict__ Bmat) {
  const int idx = blockIdx.x * blockDim.x + threadIdx.x;
  if (idx >= 2 * NB * BS * BS) return;
  const int o = idx & 127, m = (idx >> 7) & 127, mb = idx >> 14;
  const int l = mb >> 3, blk = mb & 7, fblk = (NB - blk) & 7;
  const float* t  = T + ((size_t)mb << 14) + (m << 7);
  const float* sf = S + (((size_t)((l << 3) | fblk)) << 14) + (m << 7);
  float t1 = 0.f, t2 = 0.f;
  for (int i = 0; i < BS; ++i) {
    const float h = H[i * BS + o];
    t1 += t[i] * h;
    t2 += sf[i] * h;
  }
  UH[idx] = t1 + t2;
  // m is the K index, o the N index of the effective weight matrix
  Bmat[((size_t)mb << 14) + pack_off(m, o, 4)] = (__bf16)(CHALF * (t1 - t2));
}

// A_blk = c * H @ UH   (stored packed)
__global__ void k_amat(const float* __restrict__ UH, const float* __restrict__ H,
                       __bf16* __restrict__ Amat) {
  const int idx = blockIdx.x * blockDim.x + threadIdx.x;
  if (idx >= 2 * NB * BS * BS) return;
  const int o = idx & 127, m = (idx >> 7) & 127, mb = idx >> 14;
  const float* u = UH + ((size_t)mb << 14);
  float acc = 0.f;
  for (int i = 0; i < BS; ++i) acc += H[m * BS + i] * u[(i << 7) + o];
  Amat[((size_t)mb << 14) + pack_off(m, o, 4)] = (__bf16)(CHALF * acc);
}

__global__ void k_cvt_bf16(const float* __restrict__ src, __bf16* __restrict__ dst,
                           size_t n) {
  for (size_t i = (size_t)blockIdx.x * blockDim.x + threadIdx.x; i < n;
       i += (size_t)gridDim.x * blockDim.x)
    dst[i] = (__bf16)src[i];
}

// ------------------------------- main kernels ------------------------------
// xf = xbf @ H1024. Workgroup owns one 64-col panel of H1024 (128 KB LDS);
// each of 8 waves computes 8 row-tiles of 16x64.
__global__ void k_dht1024(const __bf16* __restrict__ xb, const __bf16* __restrict__ Hp,
                          __bf16* __restrict__ xf) {
  extern __shared__ __align__(16) char smem[];
  __bf16* sH = (__bf16*)smem;                       // 4*32 tiles * 512 = 64K elems
  const int ngrp = blockIdx.x & 15;                 // 16 column groups
  const int rchunk = blockIdx.x >> 4;               // 16 chunks of 64 row-tiles
  stage_lds(sH, Hp + (size_t)ngrp * 65536, 65536 * 2, threadIdx.x, blockDim.x);
  __syncthreads();

  const int lane = threadIdx.x & 31;
  const int wv = threadIdx.x >> 5;                  // 0..7
  const int col0 = ngrp * 64;
  const int half = (lane >> 4) & 1, c = lane & 15;

  for (int r = 0; r < 8; ++r) {
    const int row0 = (rchunk * 64 + wv * 8 + r) * 16;
    v8f acc[4];
#pragma unroll
    for (int t = 0; t < 4; ++t)
#pragma unroll
      for (int g = 0; g < 8; ++g) acc[t][g] = 0.f;

    for (int kk = 0; kk < HID; kk += 32) {
      const v16bf a = load_a_frag(xb, HID, row0, kk, lane);
      const int tk = kk >> 5;
      v16bf bfr[4];
#pragma unroll
      for (int t = 0; t < 4; ++t) bfr[t] = load_b_lds(sH, t * 32 + tk, lane);
#pragma unroll
      for (int t = 0; t < 4; ++t) acc[t] = wmma_bf16(a, bfr[t], acc[t]);
    }
#pragma unroll
    for (int t = 0; t < 4; ++t)
#pragma unroll
      for (int g = 0; g < 8; ++g)
        xf[(size_t)(row0 + g + 8 * half) * HID + col0 + t * 16 + c] =
            (__bf16)acc[t][g];
  }
}

// out = act( in[b]@A_blk + in[flip b]@B_blk + bias ) ; mode 0=relu, 1=softshrink
// Workgroup owns one (blk, nh) weight panel: A+B halves staged to 32 KB LDS.
__global__ void k_blocklayer(const __bf16* __restrict__ in, const __bf16* __restrict__ Am,
                             const __bf16* __restrict__ Bm, const float* __restrict__ bias,
                             __bf16* __restrict__ outp, int mode) {
  extern __shared__ __align__(16) char smem[];
  __bf16* sA = (__bf16*)smem;        // 16 tiles * 512 = 8192 elems (16 KB)
  __bf16* sB = sA + 8192;

  const int panel = blockIdx.x & 15;                // blk*2 + nh
  const int rchunk = blockIdx.x >> 4;               // 16 chunks of 64 row-tiles
  const int blk = panel >> 1, nh = panel & 1;
  const size_t woff = ((size_t)blk << 14) + (size_t)nh * 8192;  // contiguous slab
  stage_lds(sA, Am + woff, 8192 * 2, threadIdx.x, blockDim.x);
  stage_lds(sB, Bm + woff, 8192 * 2, threadIdx.x, blockDim.x);
  __syncthreads();

  const int lane = threadIdx.x & 31;
  const int wv = threadIdx.x >> 5;
  const int wcol0 = nh * 64;
  const int half = (lane >> 4) & 1, c = lane & 15;

  float bv[4];
#pragma unroll
  for (int t = 0; t < 4; ++t) bv[t] = bias[blk * BS + wcol0 + t * 16 + c];

  for (int r = 0; r < 8; ++r) {
    const int row0 = (rchunk * 64 + wv * 8 + r) * 16;
    const int batch = row0 >> 12;                                    // row/4096
    const int frow0 = (((Btot - batch) & 3) << 12) | (row0 & 4095);  // batch flip

    v8f acc[4];
#pragma unroll
    for (int t = 0; t < 4; ++t)
#pragma unroll
      for (int g = 0; g < 8; ++g) acc[t][g] = bv[t];

#pragma unroll
    for (int kk = 0; kk < BS; kk += 32) {
      const v16bf a  = load_a_frag(in, HID, row0,  blk * BS + kk, lane);
      const v16bf af = load_a_frag(in, HID, frow0, blk * BS + kk, lane);
      const int tk = kk >> 5;
      v16bf ba[4], bb[4];
#pragma unroll
      for (int t = 0; t < 4; ++t) {
        ba[t] = load_b_lds(sA, t * 4 + tk, lane);
        bb[t] = load_b_lds(sB, t * 4 + tk, lane);
      }
#pragma unroll
      for (int t = 0; t < 4; ++t) {
        acc[t] = wmma_bf16(a,  ba[t], acc[t]);
        acc[t] = wmma_bf16(af, bb[t], acc[t]);
      }
    }

#pragma unroll
    for (int t = 0; t < 4; ++t)
#pragma unroll
      for (int g = 0; g < 8; ++g) {
        float v = acc[t][g];
        if (mode == 0) {
          v = v > 0.f ? v : 0.f;
        } else {
          v = (fabsf(v) > LAMBDA) ? (v - copysignf(LAMBDA, v)) : 0.f;
        }
        outp[(size_t)(row0 + g + 8 * half) * HID + blk * BS + wcol0 + t * 16 + c] =
            (__bf16)v;
      }
  }
}

// out = z @ (H128/2^24) per block + x   (fp32 output); Hf panel in 16 KB LDS
__global__ void k_idht_out(const __bf16* __restrict__ z, const __bf16* __restrict__ Hfp,
                           const float* __restrict__ x, float* __restrict__ outp) {
  extern __shared__ __align__(16) char smem[];
  __bf16* sH = (__bf16*)smem;                       // 16 tiles (16 KB)

  const int panel = blockIdx.x & 15;
  const int rchunk = blockIdx.x >> 4;
  const int blk = panel >> 1, nh = panel & 1;
  stage_lds(sH, Hfp + (size_t)nh * 8192, 8192 * 2, threadIdx.x, blockDim.x);
  __syncthreads();

  const int lane = threadIdx.x & 31;
  const int wv = threadIdx.x >> 5;
  const int wcol0 = nh * 64;
  const int half = (lane >> 4) & 1, c = lane & 15;

  for (int r = 0; r < 8; ++r) {
    const int row0 = (rchunk * 64 + wv * 8 + r) * 16;
    v8f acc[4];
#pragma unroll
    for (int t = 0; t < 4; ++t)
#pragma unroll
      for (int g = 0; g < 8; ++g) acc[t][g] = 0.f;

#pragma unroll
    for (int kk = 0; kk < BS; kk += 32) {
      const v16bf a = load_a_frag(z, HID, row0, blk * BS + kk, lane);
      const int tk = kk >> 5;
      v16bf bfr[4];
#pragma unroll
      for (int t = 0; t < 4; ++t) bfr[t] = load_b_lds(sH, t * 4 + tk, lane);
#pragma unroll
      for (int t = 0; t < 4; ++t) acc[t] = wmma_bf16(a, bfr[t], acc[t]);
    }
#pragma unroll
    for (int t = 0; t < 4; ++t)
#pragma unroll
      for (int g = 0; g < 8; ++g) {
        const size_t idx =
            (size_t)(row0 + g + 8 * half) * HID + blk * BS + wcol0 + t * 16 + c;
        outp[idx] = acc[t][g] + x[idx];
      }
  }
}

// ------------------------------- launcher ----------------------------------
extern "C" void kernel_launch(void* const* d_in, const int* in_sizes, int n_in,
                              void* d_out, int out_size, void* d_ws, size_t ws_size,
                              hipStream_t stream) {
  (void)in_sizes; (void)n_in; (void)out_size; (void)ws_size;
  const float* x  = (const float*)d_in[0];
  const float* w1 = (const float*)d_in[1];
  const float* b1 = (const float*)d_in[2];   // use b1[0] -> first 1024 floats
  const float* w2 = (const float*)d_in[3];
  const float* b2 = (const float*)d_in[4];
  float* out = (float*)d_out;

  char* ws = (char*)d_ws;
  size_t off = 0;
  auto alloc = [&](size_t bytes) -> char* {
    char* p = ws + off;
    off = (off + bytes + 255) & ~(size_t)255;
    return p;
  };
  __bf16* H1024 = (__bf16*)alloc((size_t)1024 * 1024 * 2);   // packed
  float*  H128f = (float*) alloc((size_t)BS * BS * 4);       // row-major
  __bf16* Hfbf  = (__bf16*)alloc((size_t)BS * BS * 2);       // packed
  float*  S     = (float*) alloc((size_t)2 * NB * BS * BS * 4);
  float*  T     = (float*) alloc((size_t)2 * NB * BS * BS * 4);
  float*  UH    = (float*) alloc((size_t)2 * NB * BS * BS * 4);
  __bf16* Amat  = (__bf16*)alloc((size_t)2 * NB * BS * BS * 2);  // packed
  __bf16* Bmat  = (__bf16*)alloc((size_t)2 * NB * BS * BS * 2);  // packed
  __bf16* buf0  = (__bf16*)alloc((size_t)ROWS * HID * 2);  // xbf, later o1
  __bf16* buf1  = (__bf16*)alloc((size_t)ROWS * HID * 2);  // xf,  later z

  const int nsmall = 2 * NB * BS * BS;  // 262144

  // --- setup: Hartley matrices + folded effective weights -------------------
  k_build_h<<<1024, 256, 0, stream>>>(H1024, H128f, Hfbf);
  k_sumw<<<(NB * BS * BS + 255) / 256, 256, 0, stream>>>(w1, S);
  k_sumw<<<(NB * BS * BS + 255) / 256, 256, 0, stream>>>(w2, S + NB * BS * BS);
  k_matT<<<nsmall / 256, 256, 0, stream>>>(S, H128f, T);
  k_vh<<<nsmall / 256, 256, 0, stream>>>(T, S, H128f, UH, Bmat);
  k_amat<<<nsmall / 256, 256, 0, stream>>>(UH, H128f, Amat);

  // --- main pipeline: 256 WGs x 8 waves x 8 row-tiles per kernel ------------
  k_cvt_bf16<<<2048, 256, 0, stream>>>(x, buf0, (size_t)ROWS * HID);
  k_dht1024<<<256, 256, 131072, stream>>>(buf0, H1024, buf1);       // xf -> buf1
  k_blocklayer<<<256, 256, 32768, stream>>>(buf1, Amat, Bmat, b1, buf0, 0);  // o1
  k_blocklayer<<<256, 256, 32768, stream>>>(buf0, Amat + NB * BS * BS,
                                            Bmat + NB * BS * BS, b2, buf1, 1);  // z
  k_idht_out<<<256, 256, 16384, stream>>>(buf1, Hfbf, x, out);
}
